// MultiHeadAttention_8598524527068
// MI455X (gfx1250) — matrix operand order repaired and checked by reading, not compiled, not run on hardware
//
#include <hip/hip_runtime.h>
#include <hip/hip_bf16.h>
#include <math.h>

#define N_B    16
#define QLEN   512
#define KLEN   512
#define EMB    512
#define HEADS  8
#define HD     64
#define P_TOT  1056   // 1024 rel positions + pad to a 16-row tile multiple
#define CSEG   544    // per-16-row-q-tile rel segment width (512 + 16 + 16)

typedef __attribute__((ext_vector_type(16))) _Float16 v16h;
typedef __attribute__((ext_vector_type(8)))  _Float16 v8h;
typedef __attribute__((ext_vector_type(8)))  float    v8f;

static __device__ __forceinline__ v16h cat8(v8h lo, v8h hi) {
  return __builtin_shufflevector(lo, hi, 0,1,2,3,4,5,6,7,8,9,10,11,12,13,14,15);
}

// A operand: 16x32 f16 tile from row-major source (ISA 7.12.2 16-bit A layout).
// lane&15 = row; lane>>4 selects K chunk {0..7,16..23} vs {8..15,24..31}.
static __device__ __forceinline__ v16h load_a(const _Float16* A, int lda) {
  const int lane = threadIdx.x & 31;
  const int r = lane & 15, g = lane >> 4;
  const _Float16* p = A + (size_t)r * lda + g * 8;
  v8h lo = *(const v8h*)(p);
  v8h hi = *(const v8h*)(p + 16);
  return cat8(lo, hi);
}

// B operand: 32x16 f16 tile; source Bc[col][k] row-major (i.e. B stored
// column-major). lane&15 = column; lane>>4 selects K chunk {0..7,16..23} vs {8..15,24..31}.
static __device__ __forceinline__ v16h load_b(const _Float16* Bc, int ldb) {
  const int lane = threadIdx.x & 31;
  const int c = lane & 15, g = lane >> 4;
  const _Float16* p = Bc + (size_t)c * ldb + g * 8;
  v8h lo = *(const v8h*)(p);
  v8h hi = *(const v8h*)(p + 16);
  return cat8(lo, hi);
}

static __device__ __forceinline__ v8f wmma_acc(v16h a, v16h b, v8f c) {
  return __builtin_amdgcn_wmma_f32_16x16x32_f16(
      /*neg_a=*/false, a, /*neg_b=*/false, b,
      /*c_mod=*/(short)0, c, /*reuse_a=*/false, /*reuse_b=*/false);
}

// Async DMA: 16B per lane, global -> LDS, tracked by ASYNCcnt (ISA 15.18.3
// opcode 98). vdst = per-lane LDS byte address, vaddr = per-lane 64b address.
static __device__ __forceinline__ void async_g2l_b128(unsigned lds_addr,
                                                      const _Float16* gptr) {
  asm volatile("global_load_async_to_lds_b128 %0, %1, off"
               :: "v"(lds_addr), "v"(gptr) : "memory");
}

static __device__ __forceinline__ void wait_async0() {
#if __has_builtin(__builtin_amdgcn_s_wait_asynccnt)
  __builtin_amdgcn_s_wait_asynccnt(0);
#else
  asm volatile("s_wait_asynccnt 0x0" ::: "memory");
#endif
}

// ---------------------------------------------------------------- converts
__global__ __launch_bounds__(256) void k_f32h(const float* __restrict__ s,
                                              _Float16* __restrict__ d, int n) {
  int i = blockIdx.x * 256 + threadIdx.x;
  if (i < n) d[i] = (_Float16)s[i];
}

// ------------------------------------------------------------- sinusoid table
__global__ __launch_bounds__(256) void k_posemb(_Float16* __restrict__ pe) {
  int i = blockIdx.x * 256 + threadIdx.x;
  if (i >= P_TOT * EMB) return;
  int p = i >> 9;
  int j = i & 511;
  int jj = j & 255;
  float freq = __expf(-(float)jj * (9.2103403719761836f / 256.0f)); // 10000^(-2j/512)
  float ang = (float)p * freq;
  pe[i] = (_Float16)((j < 256) ? __sinf(ang) : __cosf(ang));
}

// ------------------------------------------- rel[p][e] = pos_emb[p] . Wr[e]
__global__ __launch_bounds__(128) void k_rel(const _Float16* __restrict__ pe,
                                             const _Float16* __restrict__ Wr,
                                             _Float16* __restrict__ rel) {
  int rt = blockIdx.x >> 3;
  int cg = blockIdx.x & 7;
  int wave = threadIdx.x >> 5;
  int lane = threadIdx.x & 31;
  int g = lane >> 4, col = lane & 15;
  int p0 = rt * 16;
  int e0 = (cg * 4 + wave) * 16;
  v8f c = {};
  for (int k0 = 0; k0 < EMB; k0 += 32) {
    v16h a = load_a(pe + (size_t)p0 * EMB + k0, EMB);
    v16h b = load_b(Wr + (size_t)e0 * EMB + k0, EMB);   // B col e = Wr row e
    c = wmma_acc(a, b, c);
  }
#pragma unroll
  for (int r = 0; r < 8; ++r) {
    int m = r + 8 * g;
    rel[(size_t)(p0 + m) * EMB + e0 + col] = (_Float16)c[r];
  }
}

// ---------------------------- per-head q/k/v projections (+u/v bias for q)
__global__ __launch_bounds__(128) void k_proj(
    const _Float16* __restrict__ qf, const _Float16* __restrict__ kf,
    const _Float16* __restrict__ vf,
    const _Float16* __restrict__ Wq, const _Float16* __restrict__ Wk,
    const _Float16* __restrict__ Wv,
    const float* __restrict__ ub, const float* __restrict__ vb,
    _Float16* __restrict__ qu, _Float16* __restrict__ qv,
    _Float16* __restrict__ kp, _Float16* __restrict__ vT) {
  int bid = blockIdx.x;
  int pt = bid & 31;
  int h  = (bid >> 5) & 7;
  int n  = bid >> 8;
  int wave = threadIdx.x >> 5;   // 0:q+u  1:q+v  2:k  3:v(transposed)
  int lane = threadIdx.x & 31;
  int g = lane >> 4, col = lane & 15;
  int p0 = pt * 16;

  const _Float16* src = (wave < 2) ? qf : (wave == 2 ? kf : vf);
  const _Float16* W   = (wave < 2) ? Wq : (wave == 2 ? Wk : Wv);
  const _Float16* A   = src + ((size_t)n * 512 + p0) * EMB + h * HD;

  v16h a0 = load_a(A, EMB);
  v16h a1 = load_a(A + 32, EMB);
#pragma unroll
  for (int dt = 0; dt < 4; ++dt) {
    v8f c = {};
    c = wmma_acc(a0, load_b(W + (size_t)(dt * 16) * HD, HD), c);        // out[d'] = x . W[d',:]
    c = wmma_acc(a1, load_b(W + (size_t)(dt * 16) * HD + 32, HD), c);
#pragma unroll
    for (int r = 0; r < 8; ++r) {
      int m = r + 8 * g;
      int d = dt * 16 + col;
      float val = c[r];
      size_t idx = (((size_t)(n * HEADS + h)) * 512 + p0 + m) * HD + d;
      if (wave == 0)      qu[idx] = (_Float16)(val + ub[h * HD + d]);
      else if (wave == 1) qv[idx] = (_Float16)(val + vb[h * HD + d]);
      else if (wave == 2) kp[idx] = (_Float16)val;
      else vT[(((size_t)(n * HEADS + h)) * HD + d) * 512 + p0 + m] = (_Float16)val;
    }
  }
}

// ---- fused: content+position scores -> mask -> softmax -> attn out -> P@V
__global__ __launch_bounds__(128) void k_scores(
    const _Float16* __restrict__ qu, const _Float16* __restrict__ qv,
    const _Float16* __restrict__ kp, const _Float16* __restrict__ rel,
    const _Float16* __restrict__ vT, const int* __restrict__ mask,
    float* __restrict__ attn, _Float16* __restrict__ Obuf) {
  extern __shared__ char smem[];
  float*    s_cseg = (float*)smem;                   // [16][CSEG]   34816 B
  float*    s_sc   = s_cseg + 16 * CSEG;             // [16][512]    32768 B
  _Float16* s_at   = (_Float16*)(s_sc + 16 * 512);   // [16][512]    16384 B
  float*    s_red  = (float*)(s_at + 16 * 512);      // [16][8]        512 B
  _Float16* s_v    = (_Float16*)(s_red + 16 * 8);    // [64][512]    65536 B

  int bid = blockIdx.x;
  int qt = bid & 31;
  int h  = (bid >> 5) & 7;
  int n  = bid >> 8;
  int q0 = qt * 16;
  int wave = threadIdx.x >> 5;
  int lane = threadIdx.x & 31;
  int g = lane >> 4, col = lane & 15;
  int tid = threadIdx.x;

  const size_t nh = (size_t)(n * HEADS + h);
  const _Float16* quA  = qu + (nh * 512 + q0) * HD;
  const _Float16* qvA  = qv + (nh * 512 + q0) * HD;
  const _Float16* kpB  = kp + nh * 512 * HD;
  const _Float16* relB = rel + h * HD;               // rel[p] row at relB + p*EMB
  const _Float16* vTB  = vT + nh * HD * 512;

  // Step 0: kick off async DMA of the whole vT slice for this (n,h) into LDS.
  // 64x512 f16 = 64KB = 4096 x b128; overlaps with steps 1-3 (ASYNCcnt).
  {
    unsigned sv_base = (unsigned)(unsigned long long)(void*)s_v;
    for (int i = tid; i < 4096; i += 128)
      async_g2l_b128(sv_base + (unsigned)i * 16u, vTB + (size_t)i * 8);
  }

  // Step 1: position segment Cseg[m][pp] = qv[q0+m] . rel[q0+pp]
  // pos[q][k] = qv[q].rel[q+512-k]  ==>  Cseg[m][m+512-k]  (pp in [1,527])
  {
    v16h a0 = load_a(qvA, HD);
    v16h a1 = load_a(qvA + 32, HD);
    for (int pt = wave; pt < CSEG / 16; pt += 4) {
      v8f c = {};
      c = wmma_acc(a0, load_b(relB + (size_t)(q0 + pt * 16) * EMB, EMB), c);
      c = wmma_acc(a1, load_b(relB + (size_t)(q0 + pt * 16) * EMB + 32, EMB), c);
#pragma unroll
      for (int r = 0; r < 8; ++r) {
        int m = r + 8 * g;
        s_cseg[m * CSEG + pt * 16 + col] = c[r];
      }
    }
  }
  __syncthreads();

  // Step 2: energy = content + gathered position; mask; pre-scale by 1/sqrt(D)
  {
    v16h a0 = load_a(quA, HD);
    v16h a1 = load_a(quA + 32, HD);
    for (int kt = wave * 8; kt < wave * 8 + 8; ++kt) {
      if (kt < 31)  // pull next k-tile toward the WGP (global_prefetch_b8)
        __builtin_prefetch(kpB + (size_t)((kt + 1) * 16) * HD, 0, 3);
      v8f c = {};
      c = wmma_acc(a0, load_b(kpB + (size_t)(kt * 16) * HD, HD), c);
      c = wmma_acc(a1, load_b(kpB + (size_t)(kt * 16) * HD + 32, HD), c);
#pragma unroll
      for (int r = 0; r < 8; ++r) {
        int m = r + 8 * g;
        int k = kt * 16 + col;
        float e = c[r] + s_cseg[m * CSEG + (m + 512 - k)];
        int mk = mask[((size_t)n * 512 + q0 + m) * 512 + k];
        e = mk ? e : -1e20f;
        s_sc[m * 512 + k] = e * 0.125f;
      }
    }
  }
  __syncthreads();

  // Step 3: row softmax (8 threads per row), write attn (f32 output #2)
  {
    int row = tid >> 3;
    int t = tid & 7;
    float mx = -3.4e38f;
    for (int k = t; k < 512; k += 8) mx = fmaxf(mx, s_sc[row * 512 + k]);
    s_red[row * 8 + t] = mx;
    __syncthreads();
    mx = s_red[row * 8];
#pragma unroll
    for (int i = 1; i < 8; ++i) mx = fmaxf(mx, s_red[row * 8 + i]);
    __syncthreads();
    float sum = 0.f;
    for (int k = t; k < 512; k += 8) {
      float p = __expf(s_sc[row * 512 + k] - mx);
      s_sc[row * 512 + k] = p;
      sum += p;
    }
    s_red[row * 8 + t] = sum;
    __syncthreads();
    float tot = s_red[row * 8];
#pragma unroll
    for (int i = 1; i < 8; ++i) tot += s_red[row * 8 + i];
    float inv = 1.0f / tot;
    for (int k = t; k < 512; k += 8) {
      float p = s_sc[row * 512 + k] * inv;
      attn[(nh * 512 + q0 + row) * 512 + k] = p;
      s_at[row * 512 + k] = (_Float16)p;
    }
  }

  // Join the async DMA (each wave waits for its own copies, barrier makes all
  // waves' LDS writes visible), then run P@V entirely out of LDS.
  wait_async0();
  __syncthreads();

  // Step 4: O = attn @ v ; each wave owns one 16-wide slice of d, full K=512
  {
    v8f c = {};
    for (int kk = 0; kk < 512; kk += 32) {
      v16h a = load_a(s_at + kk, 512);                          // attn tile (LDS)
      v16h b = load_b(s_v + (size_t)(wave * 16) * 512 + kk, 512); // vT tile (LDS)
      c = wmma_acc(a, b, c);
    }
#pragma unroll
    for (int r = 0; r < 8; ++r) {
      int m = r + 8 * g;
      Obuf[((size_t)n * 512 + q0 + m) * EMB + h * HD + wave * 16 + col] = (_Float16)c[r];
    }
  }
}

// --------------------------------------------- out = O @ Wo^T + bo (f32 out)
__global__ __launch_bounds__(128) void k_oproj(
    const _Float16* __restrict__ O, const _Float16* __restrict__ Wo,
    const float* __restrict__ bo, float* __restrict__ out) {
  int bid = blockIdx.x;
  int rt = bid >> 3;
  int eg = bid & 7;
  int wave = threadIdx.x >> 5;
  int lane = threadIdx.x & 31;
  int g = lane >> 4, col = lane & 15;
  int row0 = rt * 16;
  int e0 = eg * 64 + wave * 16;
  v8f c = {};
  for (int k0 = 0; k0 < EMB; k0 += 32) {
    v16h a = load_a(O + (size_t)row0 * EMB + k0, EMB);
    v16h b = load_b(Wo + (size_t)e0 * EMB + k0, EMB);
    c = wmma_acc(a, b, c);
  }
#pragma unroll
  for (int r = 0; r < 8; ++r) {
    int m = r + 8 * g;
    out[(size_t)(row0 + m) * EMB + e0 + col] = c[r] + bo[e0 + col];
  }
}

extern "C" void kernel_launch(void* const* d_in, const int* in_sizes, int n_in,
                              void* d_out, int out_size, void* d_ws, size_t ws_size,
                              hipStream_t stream) {
  (void)in_sizes; (void)n_in; (void)out_size; (void)ws_size;
  const float* values = (const float*)d_in[0];
  const float* keysv  = (const float*)d_in[1];
  const float* query  = (const float*)d_in[2];
  const int*   mask   = (const int*)d_in[3];
  const float* Wv     = (const float*)d_in[4];
  const float* Wk     = (const float*)d_in[5];
  const float* Wq     = (const float*)d_in[6];
  const float* Wr     = (const float*)d_in[7];
  const float* ub     = (const float*)d_in[8];
  const float* vb     = (const float*)d_in[9];
  const float* Wo     = (const float*)d_in[10];
  const float* bo     = (const float*)d_in[11];

  float* out_main = (float*)d_out;                                   // [N,Q,E]
  float* attn_out = out_main + (size_t)N_B * QLEN * EMB;             // [N,H,Q,K]

  char* w = (char*)d_ws;
  size_t off = 0;
  auto carve = [&](size_t bytes) -> char* {
    char* p = w + off;
    off += (bytes + 255) & ~(size_t)255;
    return p;
  };
  const size_t NTOK = (size_t)N_B * 512 * EMB;                       // 4,194,304
  _Float16* qf   = (_Float16*)carve(NTOK * 2);
  _Float16* kf   = (_Float16*)carve(NTOK * 2);
  _Float16* vf   = (_Float16*)carve(NTOK * 2);
  _Float16* Wq16 = (_Float16*)carve((size_t)HD * HD * 2);
  _Float16* Wk16 = (_Float16*)carve((size_t)HD * HD * 2);
  _Float16* Wv16 = (_Float16*)carve((size_t)HD * HD * 2);
  _Float16* Wr16 = (_Float16*)carve((size_t)EMB * EMB * 2);
  _Float16* Wo16 = (_Float16*)carve((size_t)EMB * EMB * 2);
  _Float16* pe   = (_Float16*)carve((size_t)P_TOT * EMB * 2);
  _Float16* rel  = (_Float16*)carve((size_t)P_TOT * EMB * 2);
  _Float16* qu   = (_Float16*)carve(NTOK * 2);
  _Float16* qv   = (_Float16*)carve(NTOK * 2);
  _Float16* kp   = (_Float16*)carve(NTOK * 2);
  _Float16* vT   = (_Float16*)carve(NTOK * 2);
  _Float16* Obuf = (_Float16*)carve(NTOK * 2);   // total ws use ~70 MB

  auto cvt = [&](const float* s, _Float16* d, size_t n) {
    k_f32h<<<(unsigned)((n + 255) / 256), 256, 0, stream>>>(s, d, (int)n);
  };
  cvt(query, qf, NTOK);
  cvt(keysv, kf, NTOK);
  cvt(values, vf, NTOK);
  cvt(Wq, Wq16, (size_t)HD * HD);
  cvt(Wk, Wk16, (size_t)HD * HD);
  cvt(Wv, Wv16, (size_t)HD * HD);
  cvt(Wr, Wr16, (size_t)EMB * EMB);
  cvt(Wo, Wo16, (size_t)EMB * EMB);

  k_posemb<<<(P_TOT * EMB + 255) / 256, 256, 0, stream>>>(pe);
  k_rel<<<(P_TOT / 16) * 8, 128, 0, stream>>>(pe, Wr16, rel);
  k_proj<<<N_B * HEADS * 32, 128, 0, stream>>>(qf, kf, vf, Wq16, Wk16, Wv16,
                                               ub, vb, qu, qv, kp, vT);

  const size_t smem = (size_t)16 * CSEG * 4   // position segment (f32)
                    + (size_t)16 * 512  * 4   // score rows (f32)
                    + (size_t)16 * 512  * 2   // attn rows (f16)
                    + (size_t)16 * 8    * 4   // reductions
                    + (size_t)64 * 512  * 2;  // async-staged vT slice
  k_scores<<<N_B * HEADS * 32, 128, smem, stream>>>(qu, qv, kp, rel, vT, mask,
                                                    attn_out, Obuf);

  k_oproj<<<(N_B * QLEN / 16) * 8, 128, 0, stream>>>(Obuf, Wo16, bo, out_main);
}
